// RoIHeader_76897094468162
// MI455X (gfx1250) — matrix operand order repaired and checked by reading, not compiled, not run on hardware
//
#include <hip/hip_runtime.h>

#define ROI_K 300
#define CCH 512
#define FH 38
#define FW 50
#define PP 7
#define DIN (CCH * PP * PP)   // 25088
#define DH 4096
#define MPAD 384              // 300 padded to 3 blocks of 128
#define NHEAD 128             // 84 (loc) + 21 (score) padded
#define NL 84
#define NS 21

typedef __attribute__((ext_vector_type(16))) __bf16 v16bf;
typedef __attribute__((ext_vector_type(8)))  __bf16 v8bf;
typedef __attribute__((ext_vector_type(2)))  __bf16 v2bf;
typedef __attribute__((ext_vector_type(8)))  float  v8f;

// ---------------------------------------------------------------------------
// RoI max-pool (torchvision RoIPool semantics), output bf16 [MPAD, 25088].
// Blocks >= ROI_K write zero pad rows so the padded GEMM A is deterministic.
// ---------------------------------------------------------------------------
__global__ __launch_bounds__(256) void roi_pool_kernel(
    const float* __restrict__ x, const float* __restrict__ rois,
    const int* __restrict__ ridx, __bf16* __restrict__ out)
{
  const int k = blockIdx.x;
  if (k >= ROI_K) {
    for (int o = threadIdx.x; o < DIN; o += blockDim.x)
      out[(size_t)k * DIN + o] = (__bf16)0.0f;
    return;
  }
  const float scale = 1.0f / 16.0f;
  const float x1 = rois[4 * k + 0], y1 = rois[4 * k + 1];
  const float x2 = rois[4 * k + 2], y2 = rois[4 * k + 3];
  const int xs = (int)rintf(x1 * scale), ys = (int)rintf(y1 * scale);
  const int xe = (int)rintf(x2 * scale), ye = (int)rintf(y2 * scale);
  const float rw = fmaxf((float)(xe - xs + 1), 1.0f);
  const float rh = fmaxf((float)(ye - ys + 1), 1.0f);
  const float bw = rw / (float)PP, bh = rh / (float)PP;
  const float* feat = x + (size_t)ridx[k] * CCH * FH * FW;

  for (int o = threadIdx.x; o < DIN; o += blockDim.x) {
    const int c = o / (PP * PP);
    const int r = o - c * (PP * PP);
    const int ph = r / PP, pw = r - (r / PP) * PP;
    int wst = min(max((int)floorf((float)pw * bw) + xs, 0), FW);
    int wen = min(max((int)ceilf((float)(pw + 1) * bw) + xs, 0), FW);
    int hst = min(max((int)floorf((float)ph * bh) + ys, 0), FH);
    int hen = min(max((int)ceilf((float)(ph + 1) * bh) + ys, 0), FH);
    const float* fc = feat + (size_t)c * FH * FW;
    float m = -3.402823466e38f;
    for (int h = hst; h < hen; ++h)
      for (int w = wst; w < wen; ++w)
        m = fmaxf(m, fc[h * FW + w]);
    if (wst >= wen || hst >= hen) m = 0.0f;
    out[(size_t)k * DIN + o] = (__bf16)m;
  }
}

// ---------------------------------------------------------------------------
// WMMA GEMM: C[M,N] = act(A_bf16[M,K] @ B_f32[K,N] + bias)
// B converted fp32->bf16 on the fly (weight-streaming; no extra HBM pass).
// Block = 256 threads (8 wave32), BM=128 BN=64 BK=64.
// Pipelining: B double-buffered in LDS one tile (64) ahead; A fragments
// double-buffered in REGISTERS one half-tile (32) ahead, so each 4-WMMA block
// overlaps the next block's 8 global_load_b128 (partial loadcnt waits).
// Wave w: N-tile = (w&3)*16, M-tiles = (w>>2)*64 + {0,16,32,48}.
// grid.x = M/128 (fastest: same-N blocks co-run, W streams through L2 once),
// grid.y = N/64.  Requires K % 64 == 0.
// ---------------------------------------------------------------------------
template<bool RELU, bool OUTBF16>
__global__ __launch_bounds__(256) void gemm_wmma(
    const __bf16* __restrict__ A, int lda,
    const float*  __restrict__ B, int ldb,
    const float*  __restrict__ bias,
    void* __restrict__ Cout, int ldc, int K)
{
  __shared__ __bf16 Bs[2][64][72];        // [buf][n][k], padded k-stride

  const int tid  = threadIdx.x;
  const int wave = tid >> 5, lane = tid & 31;
  const int half = lane >> 4, lr = lane & 15;
  const int m0 = blockIdx.x * 128;
  const int n0 = blockIdx.y * 64;
  const int nn = (wave & 3) * 16;
  const int mw = (wave >> 2) * 64;

  // cooperative loader mapping: thread owns n = nf..nf+3, k rows {kk0, kk0+1}
  // (+32 on second pass); each global row read is 16 thr x 16 B = 256 B.
  const int kk0 = 2 * (tid >> 4);         // 0,2,..,30
  const int nf  = (tid & 15) * 4;         // 0,4,..,60
  const float* brow = B + n0 + nf;

  v8f acc[4] = {};

  const __bf16* arow[4];
#pragma unroll
  for (int i = 0; i < 4; ++i)
    arow[i] = A + (size_t)(m0 + mw + i * 16 + lr) * lda + half * 8;

  float4 r0[2], r1[2];                    // B prefetch registers (2 passes)

  auto loadB = [&](int kt) {
#pragma unroll
    for (int p = 0; p < 2; ++p) {
      const size_t kr = (size_t)(kt + kk0 + p * 32);
      r0[p] = *(const float4*)(brow + kr * ldb);
      r1[p] = *(const float4*)(brow + (kr + 1) * ldb);
    }
  };
  auto storeB = [&](int buf) {
#pragma unroll
    for (int p = 0; p < 2; ++p) {
      const int kk = kk0 + p * 32;
      v2bf q;
      q[0] = (__bf16)r0[p].x; q[1] = (__bf16)r1[p].x; *(v2bf*)&Bs[buf][nf + 0][kk] = q;
      q[0] = (__bf16)r0[p].y; q[1] = (__bf16)r1[p].y; *(v2bf*)&Bs[buf][nf + 1][kk] = q;
      q[0] = (__bf16)r0[p].z; q[1] = (__bf16)r1[p].z; *(v2bf*)&Bs[buf][nf + 2][kk] = q;
      q[0] = (__bf16)r0[p].w; q[1] = (__bf16)r1[p].w; *(v2bf*)&Bs[buf][nf + 3][kk] = q;
    }
  };
  // A fragments for one K-half into caller-owned registers:
  // e<8 -> K = koff + half*8 + e ; e>=8 -> K = koff + 16 + half*8 + (e-8)
  auto loadA = [&](v16bf (&af)[4], int koff) {
#pragma unroll
    for (int i = 0; i < 4; ++i) {
      const v8bf a0 = *(const v8bf*)(arow[i] + koff);
      const v8bf a1 = *(const v8bf*)(arow[i] + koff + 16);
#pragma unroll
      for (int e = 0; e < 8; ++e) { af[i][e] = a0[e]; af[i][8 + e] = a1[e]; }
    }
  };
  // B fragment: e<8 -> K = s*32 + half*8 + e ; e>=8 -> K = s*32 + 16 + half*8 + (e-8), N = nn+lr
  auto bfrag = [&](int buf, int s) {
    v16bf bfr;
    const v8bf b0 = *(const v8bf*)&Bs[buf][nn + lr][s * 32 + half * 8];
    const v8bf b1 = *(const v8bf*)&Bs[buf][nn + lr][s * 32 + 16 + half * 8];
#pragma unroll
    for (int e = 0; e < 8; ++e) { bfr[e] = b0[e]; bfr[8 + e] = b1[e]; }
    return bfr;
  };
  auto wmma4 = [&](const v16bf (&af)[4], v16bf bfr) {
#pragma unroll
    for (int i = 0; i < 4; ++i)
      acc[i] = __builtin_amdgcn_wmma_f32_16x16x32_bf16(
          false, af[i], false, bfr, (short)0, acc[i], false, false);
  };

  v16bf afA[4], afB[4];                   // register-pipelined A fragments

  // prologue: stage B tile 0 in LDS, A half 0 in registers
  loadB(0);
  storeB(0);
  __syncthreads();
  loadA(afA, 0);

  int buf = 0;
  for (int kt = 0; kt + 64 < K; kt += 64) {
    // cache prefetch two tiles ahead (register stage covers +64)
    if (kt + 128 < K) {
#pragma unroll
      for (int p = 0; p < 2; ++p)
        __builtin_prefetch(brow + (size_t)(kt + 128 + kk0 + p * 32) * ldb, 0, 1);
    }
    loadB(kt + 64);            // next B tile -> registers
    loadA(afB, kt + 32);       // next A half -> registers (overlaps wmma below)
    wmma4(afA, bfrag(buf, 0)); // 4 WMMAs, consume current A half
    loadA(afA, kt + 64);       // following A half (overlaps wmma below)
    wmma4(afB, bfrag(buf, 1));
    storeB(buf ^ 1);           // stage next B tile
    __syncthreads();           // single barrier per K-step
    buf ^= 1;
  }
  // epilogue tile (afA already holds its s=0 fragments)
  loadA(afB, K - 32);
  wmma4(afA, bfrag(buf, 0));
  wmma4(afB, bfrag(buf, 1));

  // D layout: lane col = lr, rows = half*8 + r (ISA 16x16 f32 C/D layout)
  const int col = n0 + nn + lr;
  const float bv = bias[col];
#pragma unroll
  for (int i = 0; i < 4; ++i) {
#pragma unroll
    for (int r = 0; r < 8; ++r) {
      const int gm = m0 + mw + i * 16 + half * 8 + r;
      float v = acc[i][r] + bv;
      if (RELU) v = fmaxf(v, 0.0f);
      if (OUTBF16) ((__bf16*)Cout)[(size_t)gm * ldc + col] = (__bf16)v;
      else         ((float*) Cout)[(size_t)gm * ldc + col] = v;
    }
  }
}

// ---------------------------------------------------------------------------
// Pack Wl [4096,84] and Ws [4096,21] into Wh [4096,128] (zero padded) + bias.
// ---------------------------------------------------------------------------
__global__ void prep_heads_kernel(const float* __restrict__ Wl, const float* __restrict__ bl,
                                  const float* __restrict__ Wsc, const float* __restrict__ bs,
                                  float* __restrict__ Wh, float* __restrict__ bh)
{
  const int i = blockIdx.x * blockDim.x + threadIdx.x;
  if (i >= DH * NHEAD) return;
  const int r = i >> 7, c = i & (NHEAD - 1);
  float v = 0.0f;
  if (c < NL)           v = Wl[r * NL + c];
  else if (c < NL + NS) v = Wsc[r * NS + (c - NL)];
  Wh[i] = v;
  if (r == 0) bh[c] = (c < NL) ? bl[c] : ((c < NL + NS) ? bs[c - NL] : 0.0f);
}

// ---------------------------------------------------------------------------
// Scatter Ch [MPAD,128] fp32 into tuple output: [300*84] locs ++ [300*21] scores
// ---------------------------------------------------------------------------
__global__ void scatter_kernel(const float* __restrict__ Ch, float* __restrict__ out)
{
  const int i = blockIdx.x * blockDim.x + threadIdx.x;
  if (i >= ROI_K * (NL + NS)) return;
  const int k = i / (NL + NS), j = i - k * (NL + NS);
  const float v = Ch[(size_t)k * NHEAD + j];
  if (j < NL) out[(size_t)k * NL + j] = v;
  else        out[(size_t)ROI_K * NL + (size_t)k * NS + (j - NL)] = v;
}

// ---------------------------------------------------------------------------
extern "C" void kernel_launch(void* const* d_in, const int* in_sizes, int n_in,
                              void* d_out, int out_size, void* d_ws, size_t ws_size,
                              hipStream_t stream)
{
  (void)in_sizes; (void)n_in; (void)out_size; (void)ws_size;
  const float* x    = (const float*)d_in[0];
  const float* rois = (const float*)d_in[1];
  const int*   ridx = (const int*)  d_in[2];
  const float* W1   = (const float*)d_in[3];
  const float* b1   = (const float*)d_in[4];
  const float* W2   = (const float*)d_in[5];
  const float* b2   = (const float*)d_in[6];
  const float* Wl   = (const float*)d_in[7];
  const float* bl   = (const float*)d_in[8];
  const float* Wsc  = (const float*)d_in[9];
  const float* bs   = (const float*)d_in[10];

  char* ws = (char*)d_ws;
  size_t off = 0;
  auto alloc = [&](size_t bytes) {
    void* p = ws + off;
    off = (off + bytes + 255) & ~(size_t)255;
    return p;
  };
  __bf16* A1 = (__bf16*)alloc((size_t)MPAD * DIN * 2);   // pooled, bf16
  __bf16* F6 = (__bf16*)alloc((size_t)MPAD * DH * 2);    // fc6, bf16
  __bf16* F7 = (__bf16*)alloc((size_t)MPAD * DH * 2);    // fc7, bf16
  float*  Wh = (float*) alloc((size_t)DH * NHEAD * 4);   // packed head weights
  float*  bh = (float*) alloc((size_t)NHEAD * 4);
  float*  Ch = (float*) alloc((size_t)MPAD * NHEAD * 4); // head GEMM result

  roi_pool_kernel<<<MPAD, 256, 0, stream>>>(x, rois, ridx, A1);

  // fc6 = relu(A1 @ W1 + b1)   M=384 N=4096 K=25088
  gemm_wmma<true, true><<<dim3(MPAD / 128, DH / 64), 256, 0, stream>>>(
      A1, DIN, W1, DH, b1, (void*)F6, DH, DIN);

  // fc7 = relu(F6 @ W2 + b2)   M=384 N=4096 K=4096
  gemm_wmma<true, true><<<dim3(MPAD / 128, DH / 64), 256, 0, stream>>>(
      F6, DH, W2, DH, b2, (void*)F7, DH, DH);

  prep_heads_kernel<<<(DH * NHEAD + 255) / 256, 256, 0, stream>>>(Wl, bl, Wsc, bs, Wh, bh);

  // heads = F7 @ Wh + bh       M=384 N=128 K=4096
  gemm_wmma<false, false><<<dim3(MPAD / 128, NHEAD / 64), 256, 0, stream>>>(
      F7, DH, Wh, NHEAD, bh, (void*)Ch, NHEAD, DH);

  scatter_kernel<<<(ROI_K * (NL + NS) + 255) / 256, 256, 0, stream>>>(Ch, (float*)d_out);
}
